// End_ExpansionNet_v2_ONNX_TensorRT_26637387170284
// MI455X (gfx1250) — matrix operand order repaired and checked by reading, not compiled, not run on hardware
//
#include <hip/hip_runtime.h>
#include <math.h>

typedef __bf16 bf16;
typedef long long i64;
typedef __attribute__((ext_vector_type(16))) __bf16 v16bf;
typedef __attribute__((ext_vector_type(8)))  __bf16 v8bf;
typedef __attribute__((ext_vector_type(8)))  float  v8f;
typedef __attribute__((ext_vector_type(4)))  unsigned int u32x4;
typedef __attribute__((ext_vector_type(8)))  int i32x8;
typedef __attribute__((ext_vector_type(4)))  int i32x4;

#define Dm   512
#define Ld   20
#define Bs   512
#define Ne   16
#define Ec   144
#define Ff   2048
#define Hh   8
#define M1v  (Bs*Ld)      /* 10240 decoder rows  */
#define MEv  (Bs*Ec)      /* 73728 encoder rows  */
#define Qn   (Ld*Ne)      /* 320 expansion rows  */
#define EPSV 1e-4f

struct GemmP {
  const bf16* A; const bf16* B;
  const float* bias; const float* resid;
  const float* expb; const float* condp; const int* nidx;
  void* out;
  i64 sA, sA2, sB, sB2, sOut, sOut2, obstride;
  float alpha;
  int M, N, K, lda, ldb, ldc;
  int relu, zdiv, rowsPer, ldt;
};

__device__ __forceinline__ int dmin(int a, int b) { return a < b ? a : b; }

__device__ __forceinline__ v16bf ld_frag(const bf16* lo, const bf16* hi) {
  v8bf a = *(const v8bf*)lo;
  v8bf b = *(const v8bf*)hi;
  return __builtin_shufflevector(a, b, 0,1,2,3,4,5,6,7,8,9,10,11,12,13,14,15);
}

// ---------------------------------------------------------------------------
// TDM: issue a 2D tile load (rows x 64B) from global into LDS.
// Descriptor per CDNA5 ISA ch.8: group0 = {count|flags, lds_addr,
// global_addr[31:0], global_addr[56:32]|type=2}; group1 = {data_size=8B,
// tensor dims (large), tile_dim0 = 8x8B = 64B, tile_dim1 = rows,
// tensor_dim0_stride = pitch in 8B units}.  Tracked by TENSORcnt.
// ---------------------------------------------------------------------------
__device__ __forceinline__ void tdm_load_2d(unsigned ldsOff, const void* gptr,
                                            unsigned rows, unsigned pitch8) {
  unsigned long long ga = (unsigned long long)gptr;
  u32x4 g0;
  g0.x = 1u;                                                 // count=1, user D#
  g0.y = ldsOff;                                             // lds_addr (bytes)
  g0.z = (unsigned)(ga & 0xffffffffu);                       // global_addr[31:0]
  g0.w = (unsigned)((ga >> 32) & 0x01ffffffu) | (2u << 30);  // addr[56:32] | type=2
  i32x8 g1;
  g1[0] = (int)(3u << 16);                       // workgroup_mask=0, data_size=3 (8B)
  g1[1] = (int)(0xffffu << 16);                  // tensor_dim0[15:0] (large)
  g1[2] = (int)(0x7fffu | (0xffffu << 16));      // tensor_dim0 hi | tensor_dim1 lo
  g1[3] = (int)(0x7fffu | (8u << 16));           // tensor_dim1 hi | tile_dim0 = 8
  g1[4] = (int)(rows & 0xffffu);                 // tile_dim1 = rows, tile_dim2 = 0
  g1[5] = (int)pitch8;                           // tensor_dim0_stride[31:0]
  g1[6] = 0;                                     // stride hi | dim1_stride lo
  g1[7] = 0;
  i32x4 z4 = {0, 0, 0, 0};
#if defined(__clang_major__) && (__clang_major__ >= 23)
  i32x8 z8 = {0, 0, 0, 0, 0, 0, 0, 0};
  __builtin_amdgcn_tensor_load_to_lds(g0, g1, z4, z4, z8, 0);
#else
  __builtin_amdgcn_tensor_load_to_lds(g0, g1, z4, z4, 0);
#endif
}

// ---------------------------------------------------------------------------
// TDM + LDS double-buffered WMMA GEMM for flat shapes (M%128==0, N%64==0).
// Wave 0 DMA's the 128x32 A tile and 64x32 B tile for step k+1 while all
// 8 waves run WMMA on step k out of LDS.
// EPI: 0 = f32 out (+opt resid), 1 = bf16 (+opt relu), 2 = bf16 transposed.
// ---------------------------------------------------------------------------
template<int EPI>
__global__ __launch_bounds__(256) void wgemm_lds(GemmP p) {
  __shared__ bf16 shA[2][128 * 32];   // row r at byte r*64
  __shared__ bf16 shB[2][64 * 32];    // col c at byte c*64
  const int wid  = threadIdx.x >> 5;
  const int lane = threadIdx.x & 31;
  const int wr = wid >> 1, wc = wid & 1;
  const int rowBlk = blockIdx.y * 128;
  const int colBlk = blockIdx.x * 64;
  const int r = lane & 15, h = lane >> 4;

  const bf16* Ab = p.A + (i64)rowBlk * p.lda;
  const bf16* Bb = p.B + (i64)colBlk * p.ldb;
  const unsigned pa8 = (unsigned)(p.lda >> 2);   // row pitch in 8B units
  const unsigned pb8 = (unsigned)(p.ldb >> 2);
  const unsigned ldsA0 = (unsigned)(unsigned long long)(const void*)&shA[0][0];
  const unsigned ldsA1 = (unsigned)(unsigned long long)(const void*)&shA[1][0];
  const unsigned ldsB0 = (unsigned)(unsigned long long)(const void*)&shB[0][0];
  const unsigned ldsB1 = (unsigned)(unsigned long long)(const void*)&shB[1][0];

  if (wid == 0) {                      // prologue: stage k=0 into buffer 0
    tdm_load_2d(ldsA0, Ab, 128, pa8);
    tdm_load_2d(ldsB0, Bb, 64,  pb8);
  }

  v8f c00{}, c01{}, c10{}, c11{};
  int par = 0;
  for (int k = 0; k < p.K; k += 32, par ^= 1) {
    if (wid == 0) __builtin_amdgcn_s_wait_tensorcnt((short)0);
    __syncthreads();                   // tile k visible to all waves
    if (wid == 0 && (k + 32) < p.K) {  // prefetch k+1 into the other buffer
      tdm_load_2d(par ? ldsA0 : ldsA1, Ab + k + 32, 128, pa8);
      tdm_load_2d(par ? ldsB0 : ldsB1, Bb + k + 32, 64,  pb8);
    }
    const bf16* sa0 = &shA[par][(wr * 32 + r)      * 32 + 8 * h];
    const bf16* sa1 = &shA[par][(wr * 32 + 16 + r) * 32 + 8 * h];
    const bf16* sb0 = &shB[par][(wc * 32 + r)      * 32 + 8 * h];
    const bf16* sb1 = &shB[par][(wc * 32 + 16 + r) * 32 + 8 * h];
    v16bf a0 = ld_frag(sa0, sa0 + 16);
    v16bf a1 = ld_frag(sa1, sa1 + 16);
    v16bf b0 = ld_frag(sb0, sb0 + 16);
    v16bf b1 = ld_frag(sb1, sb1 + 16);
    c00 = __builtin_amdgcn_wmma_f32_16x16x32_bf16(false, a0, false, b0, (short)0, c00, false, false);
    c01 = __builtin_amdgcn_wmma_f32_16x16x32_bf16(false, a0, false, b1, (short)0, c01, false, false);
    c10 = __builtin_amdgcn_wmma_f32_16x16x32_bf16(false, a1, false, b0, (short)0, c10, false, false);
    c11 = __builtin_amdgcn_wmma_f32_16x16x32_bf16(false, a1, false, b1, (short)0, c11, false, false);
  }

  const int rowBase = rowBlk + wr * 32;
  const int colBase = colBlk + wc * 32;
  const v8f accs[4] = {c00, c01, c10, c11};
  #pragma unroll
  for (int t = 0; t < 4; ++t) {
    const int rt = (t >> 1) * 16, ct = (t & 1) * 16;
    const int n = colBase + ct + (lane & 15);
    const float bv = p.bias ? p.bias[n] : 0.0f;
    #pragma unroll
    for (int i = 0; i < 8; ++i) {
      const int m = rowBase + rt + 8 * h + i;
      float v = accs[t][i] * p.alpha + bv;
      if (EPI == 0) {
        float* O = (float*)p.out;
        i64 idx = (i64)m * p.ldc + n;
        if (p.resid) v += p.resid[idx];
        O[idx] = v;
      } else if (EPI == 1) {
        if (p.relu) v = fmaxf(v, 0.0f);
        ((bf16*)p.out)[(i64)m * p.ldc + n] = (bf16)v;
      } else {
        int b2 = m / p.rowsPer, r2 = m - b2 * p.rowsPer;
        ((bf16*)p.out)[(i64)b2 * p.obstride + (i64)n * p.ldt + r2] = (bf16)v;
      }
    }
  }
}

// ---------------------------------------------------------------------------
// Generic batched WMMA GEMM (direct-from-global, clamped ragged edges) for the
// small/batched shapes: C = alpha * A[M,K] @ W[N,K]^T (+bias)(+resid).
// EPI: 0 = f32, 1 = bf16(+relu), 2 = bf16 transposed, 3 = expansion-bias + T.
// ---------------------------------------------------------------------------
template<int EPI>
__global__ __launch_bounds__(256) void wgemm(GemmP p) {
  const int z  = blockIdx.z;
  const int zb = z / p.zdiv;
  const int zh = z - zb * p.zdiv;
  const bf16* A = p.A + (i64)zb * p.sA + (i64)zh * p.sA2;
  const bf16* B = p.B + (i64)zb * p.sB + (i64)zh * p.sB2;
  const i64 outOff = (i64)zb * p.sOut + (i64)zh * p.sOut2;

  const int wid  = threadIdx.x >> 5;
  const int lane = threadIdx.x & 31;
  const int rowBase = blockIdx.y * 128 + (wid >> 1) * 32;
  const int colBase = blockIdx.x * 64  + (wid & 1) * 32;
  if (rowBase >= p.M || colBase >= p.N) return;   // wave-uniform: EXEC stays full

  const int r = lane & 15, h = lane >> 4;
  const int ar0 = dmin(rowBase + r,      p.M - 1);
  const int ar1 = dmin(rowBase + 16 + r, p.M - 1);
  const int bc0 = dmin(colBase + r,      p.N - 1);
  const int bc1 = dmin(colBase + 16 + r, p.N - 1);
  const bf16* pa0 = A + (i64)ar0 * p.lda + 8 * h;   // A frag: K halves 8h / 16+8h
  const bf16* pa1 = A + (i64)ar1 * p.lda + 8 * h;
  const bf16* pb0 = B + (i64)bc0 * p.ldb + 8 * h;   // B frag: K halves 8h / 16+8h
  const bf16* pb1 = B + (i64)bc1 * p.ldb + 8 * h;

  v8f c00{}, c01{}, c10{}, c11{};
  for (int k = 0; k < p.K; k += 32) {
    v16bf a0 = ld_frag(pa0 + k, pa0 + k + 16);
    v16bf a1 = ld_frag(pa1 + k, pa1 + k + 16);
    v16bf b0 = ld_frag(pb0 + k, pb0 + k + 16);
    v16bf b1 = ld_frag(pb1 + k, pb1 + k + 16);
    if (k + 32 < p.K) {
      __builtin_prefetch(pa0 + k + 32, 0, 1);
      __builtin_prefetch(pb0 + k + 32, 0, 1);
    }
    c00 = __builtin_amdgcn_wmma_f32_16x16x32_bf16(false, a0, false, b0, (short)0, c00, false, false);
    c01 = __builtin_amdgcn_wmma_f32_16x16x32_bf16(false, a0, false, b1, (short)0, c01, false, false);
    c10 = __builtin_amdgcn_wmma_f32_16x16x32_bf16(false, a1, false, b0, (short)0, c10, false, false);
    c11 = __builtin_amdgcn_wmma_f32_16x16x32_bf16(false, a1, false, b1, (short)0, c11, false, false);
  }

  const v8f accs[4] = {c00, c01, c10, c11};
  #pragma unroll
  for (int t = 0; t < 4; ++t) {
    const int rt = (t >> 1) * 16, ct = (t & 1) * 16;
    const int n = colBase + ct + (lane & 15);
    if (n >= p.N) continue;
    const float bv = p.bias ? p.bias[n] : 0.0f;
    #pragma unroll
    for (int i = 0; i < 8; ++i) {
      const int m = rowBase + rt + 8 * h + i;   // C layout: VGPR i -> row i+8h
      if (m >= p.M) continue;
      float v = accs[t][i] * p.alpha + bv;
      if (EPI == 0) {
        float* O = (float*)p.out;
        i64 idx = outOff + (i64)m * p.ldc + n;
        if (p.resid) v += p.resid[idx];
        O[idx] = v;
      } else if (EPI == 1) {
        if (p.relu) v = fmaxf(v, 0.0f);
        ((bf16*)p.out)[outOff + (i64)m * p.ldc + n] = (bf16)v;
      } else if (EPI == 2) {
        int b2 = m / p.rowsPer, r2 = m - b2 * p.rowsPer;
        ((bf16*)p.out)[(i64)b2 * p.obstride + (i64)n * p.ldt + r2] = (bf16)v;
      } else {  // expansion: cab[q][d] = ca + be[n_idx[b,e]] + cond[b,l]; store cab^T
        int l = m >> 4, e = m & 15;
        v += p.expb[(i64)p.nidx[zb * Ne + e] * Dm + n]
           + p.condp[((i64)zb * Ld + l) * Dm + n];
        ((bf16*)p.out)[outOff + (i64)n * p.ldt + m] = (bf16)v;
      }
    }
  }
}

// ---------------------------------------------------------------------------
// Elementwise / reduction helper kernels
// ---------------------------------------------------------------------------
__global__ __launch_bounds__(256) void cvt_kernel(const float* in, bf16* out, i64 n) {
  i64 i = (i64)blockIdx.x * 256 + threadIdx.x;
  if (i < n) out[i] = (bf16)in[i];
}

__global__ __launch_bounds__(256) void ln_kernel(const float* x, const float* g,
                                                 const float* bta, bf16* out, int R) {
  int row = blockIdx.x;
  if (row >= R) return;
  const float* xr = x + (i64)row * Dm;
  int t = threadIdx.x;
  float v0 = xr[t], v1 = xr[t + 256];
  float s = v0 + v1, q = v0 * v0 + v1 * v1;
  for (int o = 16; o > 0; o >>= 1) { s += __shfl_xor(s, o); q += __shfl_xor(q, o); }
  __shared__ float sh[18];
  int wid = t >> 5, lane = t & 31;
  if (lane == 0) { sh[wid] = s; sh[8 + wid] = q; }
  __syncthreads();
  if (t == 0) {
    float S = 0.f, Q = 0.f;
    for (int i = 0; i < 8; ++i) { S += sh[i]; Q += sh[8 + i]; }
    float m = S / (float)Dm;
    sh[16] = m;
    sh[17] = rsqrtf(Q / (float)Dm - m * m + EPSV);
  }
  __syncthreads();
  float m = sh[16], inv = sh[17];
  out[(i64)row * Dm + t]       = (bf16)((v0 - m) * inv * g[t] + bta[t]);
  out[(i64)row * Dm + t + 256] = (bf16)((v1 - m) * inv * g[t + 256] + bta[t + 256]);
}

// qcat rows 0..15 = qe[n_idx[b,e]], rows 16..35 = cond[b,l]
__global__ __launch_bounds__(256) void build_qcat(const float* qexp, const int* nidx,
                                                  const float* condP, bf16* qcat) {
  i64 i = (i64)blockIdx.x * 256 + threadIdx.x;
  if (i >= (i64)Bs * 36 * Dm) return;
  int d = (int)(i % Dm);
  int rr = (int)((i / Dm) % 36);
  int b = (int)(i / ((i64)36 * Dm));
  float v;
  if (rr < Ne) v = qexp[(i64)nidx[b * Ne + rr] * Dm + d];
  else         v = condP[((i64)b * Ld + (rr - Ne)) * Dm + d];
  qcat[i] = (bf16)v;
}

// forward masked relu-normalize: rows (b, q=(l,e)) of length 20 -> pad 32
__global__ __launch_bounds__(256) void build_fw(const float* zs, const int* fwm,
                                                bf16* afw, bf16* bfw) {
  i64 t = (i64)blockIdx.x * 256 + threadIdx.x;
  if (t >= (i64)Bs * Qn) return;
  int b = (int)(t / Qn), qq = (int)(t % Qn);
  int l = qq >> 4, e = qq & 15;
  const float* z1 = zs + ((i64)b * 36 + e) * Ld;
  const float* z2 = zs + ((i64)b * 36 + Ne + l) * Ld;
  const int* fm = fwm + t * Ld;
  float ap[Ld], bp[Ld], sa = 0.f, sb = 0.f;
  for (int k = 0; k < Ld; ++k) {
    float zz = z1[k] + z2[k];
    bool mm = (fm[k] != 0);
    float a  = (mm && zz > 0.f) ?  zz : 0.f;
    float b2 = (mm && zz < 0.f) ? -zz : 0.f;
    ap[k] = a; bp[k] = b2; sa += a; sb += b2;
  }
  float ia = 1.f / (sa + EPSV), ib = 1.f / (sb + EPSV);
  bf16* ar = afw + t * 32; bf16* br = bfw + t * 32;
  for (int k = 0; k < Ld; ++k) { ar[k] = (bf16)(ap[k] * ia); br[k] = (bf16)(bp[k] * ib); }
  for (int k = Ld; k < 32; ++k) { ar[k] = (bf16)0.f; br[k] = (bf16)0.f; }
}

// backward masked relu-normalize: rows (b,l) of length 320
__global__ __launch_bounds__(256) void build_bw(const float* zs, const int* bwm,
                                                bf16* abw, bf16* bbw) {
  i64 t = (i64)blockIdx.x * 256 + threadIdx.x;
  if (t >= (i64)M1v) return;
  int b = (int)(t / Ld), l = (int)(t % Ld);
  const int* mrow = bwm + t * Qn;
  float sa = 0.f, sb = 0.f;
  for (int q = 0; q < Qn; ++q) {
    int l2 = q >> 4, e = q & 15;
    float zz = zs[((i64)b * 36 + e) * Ld + l] + zs[((i64)b * 36 + Ne + l2) * Ld + l];
    if (mrow[q] != 0) { if (zz > 0.f) sa += zz; else sb += -zz; }
  }
  float ia = 1.f / (sa + EPSV), ib = 1.f / (sb + EPSV);
  bf16* ar = abw + t * Qn; bf16* br = bbw + t * Qn;
  for (int q = 0; q < Qn; ++q) {
    int l2 = q >> 4, e = q & 15;
    float zz = zs[((i64)b * 36 + e) * Ld + l] + zs[((i64)b * 36 + Ne + l2) * Ld + l];
    bool mm = (mrow[q] != 0);
    ar[q] = (bf16)((mm && zz > 0.f) ?  zz * ia : 0.f);
    br[q] = (bf16)((mm && zz < 0.f) ? -zz * ib : 0.f);
  }
}

__global__ __launch_bounds__(256) void combine_de(const float* x, const float* xsel,
                                                  const float* ga, const float* gb,
                                                  float* xb, i64 n) {
  i64 i = (i64)blockIdx.x * 256 + threadIdx.x;
  if (i >= n) return;
  float s = 1.f / (1.f + __expf(-xsel[i]));
  xb[i] = x[i] + s * ga[i] + (1.f - s) * gb[i];
}

// wave-per-row masked softmax over 144, write bf16 [.,160] with zero K-padding
__global__ __launch_bounds__(256) void softmax_k(const float* s, const int* cmask, bf16* att) {
  int lane = threadIdx.x & 31;
  i64 row = (i64)blockIdx.x * 8 + (threadIdx.x >> 5);   // rows = Bs*Hh*Ld
  int b = (int)(row / (Hh * Ld));
  const float* sr = s + row * Ec;
  const int* mk = cmask + (i64)b * Ec;
  float vals[5];
  float mx = -1e30f;
  #pragma unroll
  for (int j = 0; j < 5; ++j) {
    int kx = lane + 32 * j;
    float vv = -1e30f;
    if (kx < Ec) { vv = sr[kx]; if (mk[kx] == 1) vv = -1000.0f; }
    vals[j] = vv; mx = fmaxf(mx, vv);
  }
  for (int o = 16; o > 0; o >>= 1) mx = fmaxf(mx, __shfl_xor(mx, o));
  float sum = 0.f;
  #pragma unroll
  for (int j = 0; j < 5; ++j) {
    int kx = lane + 32 * j;
    float e = (kx < Ec) ? __expf(vals[j] - mx) : 0.f;
    vals[j] = e; sum += e;
  }
  for (int o = 16; o > 0; o >>= 1) sum += __shfl_xor(sum, o);
  float inv = 1.f / sum;
  #pragma unroll
  for (int j = 0; j < 5; ++j) {
    int kx = lane + 32 * j;
    att[row * 160 + kx] = (bf16)((kx < Ec) ? vals[j] * inv : 0.f);
  }
}

// ---------------------------------------------------------------------------
// Host side
// ---------------------------------------------------------------------------
static void run_gemm(int epi, const GemmP& p, int batches, hipStream_t s) {
  dim3 grid((unsigned)((p.N + 63) / 64), (unsigned)((p.M + 127) / 128), (unsigned)batches);
  dim3 blk(256);
  switch (epi) {
    case 0: wgemm<0><<<grid, blk, 0, s>>>(p); break;
    case 1: wgemm<1><<<grid, blk, 0, s>>>(p); break;
    case 2: wgemm<2><<<grid, blk, 0, s>>>(p); break;
    default: wgemm<3><<<grid, blk, 0, s>>>(p); break;
  }
}

// flat shapes only: M%128==0, N%64==0, K%32==0
static void run_gemm_lds(int epi, const GemmP& p, hipStream_t s) {
  dim3 grid((unsigned)(p.N / 64), (unsigned)(p.M / 128), 1);
  dim3 blk(256);
  switch (epi) {
    case 0: wgemm_lds<0><<<grid, blk, 0, s>>>(p); break;
    case 1: wgemm_lds<1><<<grid, blk, 0, s>>>(p); break;
    default: wgemm_lds<2><<<grid, blk, 0, s>>>(p); break;
  }
}

static GemmP mk() { GemmP p{}; p.alpha = 1.f; p.zdiv = 1; return p; }

extern "C" void kernel_launch(void* const* d_in, const int* in_sizes, int n_in,
                              void* d_out, int out_size, void* d_ws, size_t ws_size,
                              hipStream_t stream) {
  (void)in_sizes; (void)n_in; (void)out_size; (void)ws_size;
  const float* x     = (const float*)d_in[0];
  const float* cx    = (const float*)d_in[1];
  const int*   nidx  = (const int*)  d_in[2];
  const int*   fwm   = (const int*)  d_in[3];
  const int*   bwm   = (const int*)  d_in[4];
  const int*   cmask = (const int*)  d_in[5];
  const float* ln1g = (const float*)d_in[6];  const float* ln1b = (const float*)d_in[7];
  const float* ln2g = (const float*)d_in[8];  const float* ln2b = (const float*)d_in[9];
  const float* ln3g = (const float*)d_in[10]; const float* ln3b = (const float*)d_in[11];
  const float* condw = (const float*)d_in[12]; const float* condb = (const float*)d_in[13];
  const float* qexp  = (const float*)d_in[14]; const float* bexp  = (const float*)d_in[15];
  const float* keyw  = (const float*)d_in[16]; const float* keyb  = (const float*)d_in[17];
  const float* aw    = (const float*)d_in[18]; const float* ab    = (const float*)d_in[19];
  const float* bw_   = (const float*)d_in[20]; const float* bb_   = (const float*)d_in[21];
  const float* selw  = (const float*)d_in[22]; const float* selb  = (const float*)d_in[23];
  const float* wq    = (const float*)d_in[24]; const float* wqb   = (const float*)d_in[25];
  const float* wk    = (const float*)d_in[26]; const float* wkb   = (const float*)d_in[27];
  const float* wv    = (const float*)d_in[28]; const float* wvb   = (const float*)d_in[29];
  const float* wo    = (const float*)d_in[30]; const float* wob   = (const float*)d_in[31];
  const float* f1w   = (const float*)d_in[32]; const float* f1b   = (const float*)d_in[33];
  const float* f2w   = (const float*)d_in[34]; const float* f2b   = (const float*)d_in[35];

  char* base = (char*)d_ws;
  size_t off = 0;
  auto alloc = [&](size_t bytes) -> void* {
    void* p = base + off;
    off += (bytes + 255) & ~(size_t)255;
    return p;
  };

  // ---- persistent region (live across both phases), ~120 MB ----
  bf16* Wcond = (bf16*)alloc((size_t)Dm * Dm * 2);
  bf16* Wkey  = (bf16*)alloc((size_t)Dm * Dm * 2);
  bf16* Wa    = (bf16*)alloc((size_t)Dm * Dm * 2);
  bf16* Wb    = (bf16*)alloc((size_t)Dm * Dm * 2);
  bf16* Wsel  = (bf16*)alloc((size_t)Dm * Dm * 2);
  bf16* Wq    = (bf16*)alloc((size_t)Dm * Dm * 2);
  bf16* Wk    = (bf16*)alloc((size_t)Dm * Dm * 2);
  bf16* Wv    = (bf16*)alloc((size_t)Dm * Dm * 2);
  bf16* Wo    = (bf16*)alloc((size_t)Dm * Dm * 2);
  bf16* Wff1  = (bf16*)alloc((size_t)Ff * Dm * 2);
  bf16* Wff2  = (bf16*)alloc((size_t)Dm * Ff * 2);
  bf16* cxb   = (bf16*)alloc((size_t)MEv * Dm * 2);
  bf16* x2a   = (bf16*)alloc((size_t)M1v * Dm * 2);
  float* xbuf = (float*)alloc((size_t)M1v * Dm * 4);
  size_t pbase = off;

  // ---- phase 1 scratch (DynamicExpansionBlock), ~520 MB ----
  float* condP = (float*)alloc((size_t)M1v * Dm * 4);
  bf16*  xkb   = (bf16*) alloc((size_t)M1v * Dm * 2);
  bf16*  xaT   = (bf16*) alloc((size_t)Bs * Dm * 32 * 2);
  bf16*  xbT   = (bf16*) alloc((size_t)Bs * Dm * 32 * 2);
  float* xselP = (float*)alloc((size_t)M1v * Dm * 4);
  bf16*  qcat  = (bf16*) alloc((size_t)Bs * 36 * Dm * 2);
  float* zs    = (float*)alloc((size_t)Bs * 36 * Ld * 4);
  bf16*  afw   = (bf16*) alloc((size_t)Bs * Qn * 32 * 2);
  bf16*  bfw   = (bf16*) alloc((size_t)Bs * Qn * 32 * 2);
  bf16*  abw   = (bf16*) alloc((size_t)Bs * Ld * Qn * 2);
  bf16*  bbw   = (bf16*) alloc((size_t)Bs * Ld * Qn * 2);
  bf16*  cabA  = (bf16*) alloc((size_t)Bs * Dm * Qn * 2);
  bf16*  cabB  = (bf16*) alloc((size_t)Bs * Dm * Qn * 2);
  float* ga    = (float*)alloc((size_t)M1v * Dm * 4);
  float* gb    = (float*)alloc((size_t)M1v * Dm * 4);

  auto cvtl = [&](const float* s, bf16* d, i64 n) {
    cvt_kernel<<<(unsigned)((n + 255) / 256), 256, 0, stream>>>(s, d, n);
  };
  cvtl(condw, Wcond, (i64)Dm * Dm); cvtl(keyw, Wkey, (i64)Dm * Dm);
  cvtl(aw, Wa, (i64)Dm * Dm);       cvtl(bw_, Wb, (i64)Dm * Dm);
  cvtl(selw, Wsel, (i64)Dm * Dm);   cvtl(wq, Wq, (i64)Dm * Dm);
  cvtl(wk, Wk, (i64)Dm * Dm);       cvtl(wv, Wv, (i64)Dm * Dm);
  cvtl(wo, Wo, (i64)Dm * Dm);       cvtl(f1w, Wff1, (i64)Ff * Dm);
  cvtl(f2w, Wff2, (i64)Dm * Ff);    cvtl(cx, cxb, (i64)MEv * Dm);

  // ---- DynamicExpansionBlock ----
  ln_kernel<<<M1v, 256, 0, stream>>>(x, ln1g, ln1b, x2a, M1v);

  { GemmP p = mk(); p.A = x2a; p.lda = Dm; p.K = Dm; p.M = M1v;
    p.B = Wcond; p.ldb = Dm; p.N = Dm; p.bias = condb; p.out = condP; p.ldc = Dm;
    run_gemm_lds(0, p, stream); }
  { GemmP p = mk(); p.A = x2a; p.lda = Dm; p.K = Dm; p.M = M1v;
    p.B = Wkey; p.ldb = Dm; p.N = Dm; p.bias = keyb; p.out = xkb; p.ldc = Dm;
    run_gemm_lds(1, p, stream); }
  { GemmP p = mk(); p.A = x2a; p.lda = Dm; p.K = Dm; p.M = M1v;
    p.B = Wa; p.ldb = Dm; p.N = Dm; p.bias = ab; p.out = xaT;
    p.rowsPer = Ld; p.ldt = 32; p.obstride = (i64)Dm * 32;
    run_gemm_lds(2, p, stream); }
  { GemmP p = mk(); p.A = x2a; p.lda = Dm; p.K = Dm; p.M = M1v;
    p.B = Wb; p.ldb = Dm; p.N = Dm; p.bias = bb_; p.out = xbT;
    p.rowsPer = Ld; p.ldt = 32; p.obstride = (i64)Dm * 32;
    run_gemm_lds(2, p, stream); }
  { GemmP p = mk(); p.A = x2a; p.lda = Dm; p.K = Dm; p.M = M1v;
    p.B = Wsel; p.ldb = Dm; p.N = Dm; p.bias = selb; p.out = xselP; p.ldc = Dm;
    run_gemm_lds(0, p, stream); }

  build_qcat<<<(unsigned)(((i64)Bs * 36 * Dm + 255) / 256), 256, 0, stream>>>(qexp, nidx, condP, qcat);

  // z decomposition: [36,512] x [20,512]^T per batch; z[(l,e),k] = Z[e,k] + Z[16+l,k]
  { GemmP p = mk(); p.A = qcat; p.lda = Dm; p.sA = (i64)36 * Dm;
    p.B = xkb; p.ldb = Dm; p.sB = (i64)Ld * Dm;
    p.M = 36; p.N = Ld; p.K = Dm; p.out = zs; p.ldc = Ld; p.sOut = (i64)36 * Ld;
    p.alpha = 1.0f / sqrtf((float)Dm);
    run_gemm(0, p, Bs, stream); }

  build_fw<<<(unsigned)(((i64)Bs * Qn + 255) / 256), 256, 0, stream>>>(zs, fwm, afw, bfw);
  build_bw<<<(unsigned)((M1v + 255) / 256), 256, 0, stream>>>(zs, bwm, abw, bbw);

  // cab^T = (a_fw @ xa + be[n]+cond)^T   (K=32, A zero-padded)
  { GemmP p = mk(); p.A = afw; p.lda = 32; p.sA = (i64)Qn * 32;
    p.B = xaT; p.ldb = 32; p.sB = (i64)Dm * 32;
    p.M = Qn; p.N = Dm; p.K = 32; p.out = cabA; p.sOut = (i64)Dm * Qn; p.ldt = Qn;
    p.expb = bexp; p.condp = condP; p.nidx = nidx;
    run_gemm(3, p, Bs, stream); }
  { GemmP p = mk(); p.A = bfw; p.lda = 32; p.sA = (i64)Qn * 32;
    p.B = xbT; p.ldb = 32; p.sB = (i64)Dm * 32;
    p.M = Qn; p.N = Dm; p.K = 32; p.out = cabB; p.sOut = (i64)Dm * Qn; p.ldt = Qn;
    p.expb = bexp; p.condp = condP; p.nidx = nidx;
    run_gemm(3, p, Bs, stream); }

  // g = a_bw @ cab   ([20,320] x [512,320]^T per batch)
  { GemmP p = mk(); p.A = abw; p.lda = Qn; p.sA = (i64)Ld * Qn;
    p.B = cabA; p.ldb = Qn; p.sB = (i64)Dm * Qn;
    p.M = Ld; p.N = Dm; p.K = Qn; p.out = ga; p.ldc = Dm; p.sOut = (i64)Ld * Dm;
    run_gemm(0, p, Bs, stream); }
  { GemmP p = mk(); p.A = bbw; p.lda = Qn; p.sA = (i64)Ld * Qn;
    p.B = cabB; p.ldb = Qn; p.sB = (i64)Dm * Qn;
    p.M = Ld; p.N = Dm; p.K = Qn; p.out = gb; p.ldc = Dm; p.sOut = (i64)Ld * Dm;
    run_gemm(0, p, Bs, stream); }

  combine_de<<<(unsigned)(((i64)M1v * Dm + 255) / 256), 256, 0, stream>>>(
      x, xselP, ga, gb, xbuf, (i64)M1v * Dm);

  // ---- phase 2 scratch reuses the phase-1 region (stream-ordered), ~296 MB ----
  off = pbase;
  bf16*  qb  = (bf16*) alloc((size_t)M1v * Dm * 2);
  bf16*  kb  = (bf16*) alloc((size_t)MEv * Dm * 2);
  bf16*  vT  = (bf16*) alloc((size_t)Bs * Dm * 160 * 2);
  float* sb  = (float*)alloc((size_t)Bs * Hh * Ld * Ec * 4);
  bf16*  att = (bf16*) alloc((size_t)Bs * Hh * Ld * 160 * 2);
  bf16*  ob  = (bf16*) alloc((size_t)M1v * Dm * 2);
  bf16*  h1  = (bf16*) alloc((size_t)M1v * Ff * 2);

  // ---- cross MHA ----
  ln_kernel<<<M1v, 256, 0, stream>>>(xbuf, ln2g, ln2b, x2a, M1v);

  { GemmP p = mk(); p.A = x2a; p.lda = Dm; p.K = Dm; p.M = M1v;
    p.B = Wq; p.ldb = Dm; p.N = Dm; p.bias = wqb; p.out = qb; p.ldc = Dm;
    run_gemm_lds(1, p, stream); }
  { GemmP p = mk(); p.A = cxb; p.lda = Dm; p.K = Dm; p.M = MEv;
    p.B = Wk; p.ldb = Dm; p.N = Dm; p.bias = wkb; p.out = kb; p.ldc = Dm;
    run_gemm_lds(1, p, stream); }
  { GemmP p = mk(); p.A = cxb; p.lda = Dm; p.K = Dm; p.M = MEv;
    p.B = Wv; p.ldb = Dm; p.N = Dm; p.bias = wvb; p.out = vT;
    p.rowsPer = Ec; p.ldt = 160; p.obstride = (i64)Dm * 160;
    run_gemm_lds(2, p, stream); }

  // scores: per (b,h) [20,64] x [144,64]^T, alpha = 1/sqrt(64)
  { GemmP p = mk(); p.zdiv = Hh;
    p.A = qb; p.lda = Dm; p.sA = (i64)Ld * Dm; p.sA2 = 64;
    p.B = kb; p.ldb = Dm; p.sB = (i64)Ec * Dm; p.sB2 = 64;
    p.M = Ld; p.N = Ec; p.K = 64; p.out = sb; p.ldc = Ec;
    p.sOut = (i64)Hh * Ld * Ec; p.sOut2 = (i64)Ld * Ec; p.alpha = 0.125f;
    run_gemm(0, p, Bs * Hh, stream); }

  softmax_k<<<(unsigned)(Bs * Hh * Ld / 8), 256, 0, stream>>>(sb, cmask, att);

  // out: per (b,h) [20,160] x [64,160]^T (att K-padded w/ zeros; vT garbage pad x 0)
  { GemmP p = mk(); p.zdiv = Hh;
    p.A = att; p.lda = 160; p.sA = (i64)Hh * Ld * 160; p.sA2 = (i64)Ld * 160;
    p.B = vT; p.ldb = 160; p.sB = (i64)Dm * 160; p.sB2 = (i64)64 * 160;
    p.M = Ld; p.N = 64; p.K = 160; p.out = ob; p.ldc = Dm;
    p.sOut = (i64)Ld * Dm; p.sOut2 = 64;
    run_gemm(1, p, Bs * Hh, stream); }

  { GemmP p = mk(); p.A = ob; p.lda = Dm; p.K = Dm; p.M = M1v;
    p.B = Wo; p.ldb = Dm; p.N = Dm; p.bias = wob;
    p.resid = xbuf; p.out = xbuf; p.ldc = Dm;
    run_gemm_lds(0, p, stream); }

  // ---- FeedForward ----
  ln_kernel<<<M1v, 256, 0, stream>>>(xbuf, ln3g, ln3b, x2a, M1v);

  { GemmP p = mk(); p.A = x2a; p.lda = Dm; p.K = Dm; p.M = M1v;
    p.B = Wff1; p.ldb = Dm; p.N = Ff; p.bias = f1b; p.relu = 1;
    p.out = h1; p.ldc = Ff;
    run_gemm_lds(1, p, stream); }
  { GemmP p = mk(); p.A = h1; p.lda = Ff; p.K = Ff; p.M = M1v;
    p.B = Wff2; p.ldb = Ff; p.N = Dm; p.bias = f2b;
    p.resid = xbuf; p.out = (float*)d_out; p.ldc = Dm;
    run_gemm_lds(0, p, stream); }
}
